// DoubleXLSTMUp_58557584113769
// MI455X (gfx1250) — compile-verified, matrix operand order checked by reading, not run on hardware
//
#include <hip/hip_runtime.h>
#include <math.h>

// ---------------------------------------------------------------------------
// Types / WMMA helpers (CDNA5 gfx1250, wave32)
// ---------------------------------------------------------------------------
typedef __attribute__((ext_vector_type(16))) __bf16 v16bf;
typedef __attribute__((ext_vector_type(8)))  __bf16 v8bf;
typedef __attribute__((ext_vector_type(4)))  __bf16 v4bf;
typedef __attribute__((ext_vector_type(8)))  float  v8f;

#define DEVI __device__ __forceinline__

DEVI v8f wmma_bf16(v16bf a, v16bf b, v8f c) {
  // D = A(16x32 bf16) * B(32x16 bf16) + C(16x16 f32)
  return __builtin_amdgcn_wmma_f32_16x16x32_bf16(
      /*neg_a=*/false, a, /*neg_b=*/false, b,
      /*c_mod=*/(short)0, c, /*reuse_a=*/false, /*reuse_b=*/false);
}

DEVI v8f v8f_zero() {
  v8f z;
#pragma unroll
  for (int i = 0; i < 8; ++i) z[i] = 0.0f;
  return z;
}

DEVI v16bf combine16(v8bf lo, v8bf hi) {
  return __builtin_shufflevector(lo, hi, 0, 1, 2, 3, 4, 5, 6, 7, 8, 9, 10, 11,
                                 12, 13, 14, 15);
}

// Fragment from two contiguous 16B runs in LDS (16B-aligned): p[0..7],p[16..23]
DEVI v16bf frag_lds(const __bf16* p) {
  v8bf lo = *(const v8bf*)(p);
  v8bf hi = *(const v8bf*)(p + 16);
  return combine16(lo, hi);
}

// Fragment from global f32: elements p[0..7] and p[16..23] (16B-aligned)
DEVI v16bf frag_gf32(const float* p) {
  float4 a = *(const float4*)(p);
  float4 b = *(const float4*)(p + 4);
  float4 c = *(const float4*)(p + 16);
  float4 d = *(const float4*)(p + 20);
  v16bf r;
  r[0] = (__bf16)a.x;  r[1] = (__bf16)a.y;  r[2] = (__bf16)a.z;  r[3] = (__bf16)a.w;
  r[4] = (__bf16)b.x;  r[5] = (__bf16)b.y;  r[6] = (__bf16)b.z;  r[7] = (__bf16)b.w;
  r[8] = (__bf16)c.x;  r[9] = (__bf16)c.y;  r[10] = (__bf16)c.z; r[11] = (__bf16)c.w;
  r[12] = (__bf16)d.x; r[13] = (__bf16)d.y; r[14] = (__bf16)d.z; r[15] = (__bf16)d.w;
  return r;
}

DEVI v16bf frag_gf32_scaled(const float* p, float s) {
  float4 a = *(const float4*)(p);
  float4 b = *(const float4*)(p + 4);
  float4 c = *(const float4*)(p + 16);
  float4 d = *(const float4*)(p + 20);
  v16bf r;
  r[0] = (__bf16)(a.x * s);  r[1] = (__bf16)(a.y * s);
  r[2] = (__bf16)(a.z * s);  r[3] = (__bf16)(a.w * s);
  r[4] = (__bf16)(b.x * s);  r[5] = (__bf16)(b.y * s);
  r[6] = (__bf16)(b.z * s);  r[7] = (__bf16)(b.w * s);
  r[8] = (__bf16)(c.x * s);  r[9] = (__bf16)(c.y * s);
  r[10] = (__bf16)(c.z * s); r[11] = (__bf16)(c.w * s);
  r[12] = (__bf16)(d.x * s); r[13] = (__bf16)(d.y * s);
  r[14] = (__bf16)(d.z * s); r[15] = (__bf16)(d.w * s);
  return r;
}

DEVI float wave_red_sum32(float v) {
#pragma unroll
  for (int d = 16; d > 0; d >>= 1) v += __shfl_xor(v, d, 32);
  return v;
}

// ---------------------------------------------------------------------------
// Generic f32-in / f32-out GEMM via bf16 WMMA, register-staged double buffer.
// C[m,n] = sum_k A[m,k]*B[k,n] (+bias[n]) (+add[m,n])
// Block 256 threads = 8 waves, tile 128(M) x 128(N), K-step 32.
// A tile: row-major [128][40] (80B rows, 16B aligned -> ds_load_b128 frags).
// B tile: TRANSPOSED [col][k] = [128][40] so B fragments are contiguous too.
// Pipeline: issue next tile's global loads -> frag ds loads + 8 WMMAs on the
// current buffer -> convert+store staged regs to the other buffer -> barrier.
// ---------------------------------------------------------------------------
__global__ __launch_bounds__(256) void k_gemm(
    const float* __restrict__ A, int lda,
    const float* __restrict__ Bw, int ldb,
    const float* __restrict__ bias,
    const float* add, int ldadd,
    float* C, int ldc, int Kd) {
  __shared__ __bf16 As[2][128][40];
  __shared__ __bf16 Bt[2][128][40];

  const int tid = threadIdx.x;
  const int lane = tid & 31;
  const int wid = tid >> 5;
  const int wm = wid & 3;   // 0..3 -> 32 rows each
  const int wn = wid >> 2;  // 0..1 -> 64 cols each
  const int row0 = blockIdx.y * 128;
  const int col0 = blockIdx.x * 128;
  const int cl = lane & 15;
  const int kb = (lane >> 4) << 3;  // 0 or 8 (wmma K lane split)

  // per-thread staging coordinates (4 chunks each for A and B per tile)
  const int arow = tid >> 3;         // + it*32
  const int ac4 = (tid & 7) << 2;    // 0..28
  const int brow = tid >> 5;         // + it*8   (k index)
  const int bc4 = (tid & 31) << 2;   // 0..124   (col index)

  v8f acc[2][4];
#pragma unroll
  for (int i = 0; i < 2; ++i)
#pragma unroll
    for (int j = 0; j < 4; ++j) acc[i][j] = v8f_zero();

  // ---- prologue: stage tile 0 into buffer 0 ----
  {
    float4 fa[4], fb[4];
#pragma unroll
    for (int it = 0; it < 4; ++it) {
      fa[it] = *(const float4*)&A[(row0 + arow + it * 32) * lda + ac4];
      fb[it] = *(const float4*)&Bw[(brow + it * 8) * ldb + col0 + bc4];
    }
#pragma unroll
    for (int it = 0; it < 4; ++it) {
      v4bf o;
      o[0] = (__bf16)fa[it].x; o[1] = (__bf16)fa[it].y;
      o[2] = (__bf16)fa[it].z; o[3] = (__bf16)fa[it].w;
      *(v4bf*)&As[0][arow + it * 32][ac4] = o;
      Bt[0][bc4 + 0][brow + it * 8] = (__bf16)fb[it].x;
      Bt[0][bc4 + 1][brow + it * 8] = (__bf16)fb[it].y;
      Bt[0][bc4 + 2][brow + it * 8] = (__bf16)fb[it].z;
      Bt[0][bc4 + 3][brow + it * 8] = (__bf16)fb[it].w;
    }
  }
  __syncthreads();

  const int nk = Kd >> 5;
  for (int ki = 0; ki < nk; ++ki) {
    const int cur = ki & 1;
    const bool more = (ki + 1) < nk;

    // issue next tile's global loads early (latency hides behind WMMAs)
    float4 fa[4], fb[4];
    if (more) {
      const int k0 = (ki + 1) << 5;
#pragma unroll
      for (int it = 0; it < 4; ++it) {
        fa[it] = *(const float4*)&A[(row0 + arow + it * 32) * lda + k0 + ac4];
        fb[it] = *(const float4*)&Bw[(k0 + brow + it * 8) * ldb + col0 + bc4];
      }
    }

    // fragments from current buffer + 8 WMMAs
    v16bf af[2], bfm[4];
#pragma unroll
    for (int i = 0; i < 2; ++i)
      af[i] = frag_lds(&As[cur][wm * 32 + i * 16 + cl][kb]);
#pragma unroll
    for (int j2 = 0; j2 < 4; ++j2)
      bfm[j2] = frag_lds(&Bt[cur][wn * 64 + j2 * 16 + cl][kb]);
#pragma unroll
    for (int i = 0; i < 2; ++i)
#pragma unroll
      for (int j2 = 0; j2 < 4; ++j2)
        acc[i][j2] = wmma_bf16(af[i], bfm[j2], acc[i][j2]);

    // convert + store staged registers into the other buffer
    if (more) {
      const int nb = cur ^ 1;
#pragma unroll
      for (int it = 0; it < 4; ++it) {
        v4bf o;
        o[0] = (__bf16)fa[it].x; o[1] = (__bf16)fa[it].y;
        o[2] = (__bf16)fa[it].z; o[3] = (__bf16)fa[it].w;
        *(v4bf*)&As[nb][arow + it * 32][ac4] = o;
        Bt[nb][bc4 + 0][brow + it * 8] = (__bf16)fb[it].x;
        Bt[nb][bc4 + 1][brow + it * 8] = (__bf16)fb[it].y;
        Bt[nb][bc4 + 2][brow + it * 8] = (__bf16)fb[it].z;
        Bt[nb][bc4 + 3][brow + it * 8] = (__bf16)fb[it].w;
      }
    }
    __syncthreads();
  }

  const int off = (lane >> 4) << 3;  // row offset of this half-wave in D tile
#pragma unroll
  for (int i = 0; i < 2; ++i)
#pragma unroll
    for (int j2 = 0; j2 < 4; ++j2) {
      int n = col0 + wn * 64 + j2 * 16 + cl;
      float bv = bias ? bias[n] : 0.0f;
#pragma unroll
      for (int r = 0; r < 8; ++r) {
        int m = row0 + wm * 32 + i * 16 + off + r;
        float val = acc[i][j2][r] + bv;
        if (add) val += add[m * ldadd + n];
        C[m * ldc + n] = val;
      }
    }
}

// ---------------------------------------------------------------------------
// LayerNorm over last dim 512. One wave per row, 8 rows per block.
// ---------------------------------------------------------------------------
__global__ __launch_bounds__(256) void k_ln(const float* __restrict__ x,
                                            const float* __restrict__ g,
                                            const float* __restrict__ b,
                                            float* __restrict__ out) {
  const int row = blockIdx.x * 8 + (threadIdx.x >> 5);
  const int lane = threadIdx.x & 31;
  const float* xp = x + row * 512;
  float vals[16];
  float s = 0.0f;
#pragma unroll
  for (int i = 0; i < 16; ++i) {
    vals[i] = xp[lane + i * 32];
    s += vals[i];
  }
  s = wave_red_sum32(s);
  float mu = s * (1.0f / 512.0f);
  float vs = 0.0f;
#pragma unroll
  for (int i = 0; i < 16; ++i) {
    float d = vals[i] - mu;
    vs += d * d;
  }
  vs = wave_red_sum32(vs);
  float rr = rsqrtf(vs * (1.0f / 512.0f) + 1e-6f);
#pragma unroll
  for (int i = 0; i < 16; ++i) {
    int c = lane + i * 32;
    out[row * 512 + c] = (vals[i] - mu) * rr * g[c] + b[c];
  }
}

// ---------------------------------------------------------------------------
// Causal depthwise conv (K=4) + SiLU. Elementwise over (B*S) x 1024.
// ---------------------------------------------------------------------------
__global__ __launch_bounds__(256) void k_conv_silu(
    const float* __restrict__ xin, int ldx, const float* __restrict__ w,
    const float* __restrict__ bc, float* __restrict__ xc) {
  const int idx = blockIdx.x * 256 + threadIdx.x;  // over 8192*1024
  const int c = idx & 1023;
  const int row = idx >> 10;
  const int t = row & 2047;
  float a = bc[c];
#pragma unroll
  for (int kk = 0; kk < 4; ++kk) {
    int dt = t - 3 + kk;
    if (dt >= 0) a += w[kk * 1024 + c] * xin[(row - 3 + kk) * ldx + c];
  }
  xc[idx] = a / (1.0f + __expf(-a));
}

// ---------------------------------------------------------------------------
// i/f head projections: N=4 each, K=1024. One wave per row.
// ---------------------------------------------------------------------------
__global__ __launch_bounds__(256) void k_ifproj(
    const float* __restrict__ xc, const float* __restrict__ Wi,
    const float* __restrict__ bi, const float* __restrict__ Wf,
    const float* __restrict__ bfp, float* __restrict__ iraw,
    float* __restrict__ fraw) {
  const int row = blockIdx.x * 8 + (threadIdx.x >> 5);
  const int lane = threadIdx.x & 31;
  float si[4] = {0, 0, 0, 0}, sf[4] = {0, 0, 0, 0};
  const float* xp = xc + row * 1024;
  for (int kk = lane; kk < 1024; kk += 32) {
    float xv = xp[kk];
#pragma unroll
    for (int o = 0; o < 4; ++o) {
      si[o] += xv * Wi[kk * 4 + o];
      sf[o] += xv * Wf[kk * 4 + o];
    }
  }
#pragma unroll
  for (int o = 0; o < 4; ++o) {
    si[o] = wave_red_sum32(si[o]);
    sf[o] = wave_red_sum32(sf[o]);
  }
  if (lane == 0) {
#pragma unroll
    for (int o = 0; o < 4; ++o) {
      iraw[row * 4 + o] = si[o] + bi[o];
      fraw[row * 4 + o] = sf[o] + bfp[o];
    }
  }
}

// ---------------------------------------------------------------------------
// Per-(b,h) scan: Fc = cumsum(logsigmoid(f)), G = i - Fc.
// One wave per (b,h). grid = 16 blocks of 32 threads.
// ---------------------------------------------------------------------------
__global__ void k_scan(const float* __restrict__ iraw,
                       const float* __restrict__ fraw,
                       float* __restrict__ Fc, float* __restrict__ G) {
  const int bh = blockIdx.x;
  const int b = bh >> 2, h = bh & 3;
  const int lane = threadIdx.x;
  float carry = 0.0f;
  for (int t0 = 0; t0 < 2048; t0 += 32) {
    int row = b * 2048 + t0 + lane;
    float f = fraw[row * 4 + h];
    float lf = fminf(f, 0.0f) - log1pf(__expf(-fabsf(f)));
    float val = lf;
#pragma unroll
    for (int d = 1; d < 32; d <<= 1) {
      float nb = __shfl_up(val, d, 32);
      if (lane >= d) val += nb;
    }
    float fc = carry + val;
    Fc[bh * 2048 + t0 + lane] = fc;
    G[bh * 2048 + t0 + lane] = iraw[row * 4 + h] - fc;
    carry += __shfl(val, 31, 32);
  }
}

// ---------------------------------------------------------------------------
// Parallel mLSTM, flash-style streaming. One wave per 16 query rows.
// grid: x = S/64, y = B*H ; block = 128 threads (4 waves).
// q is scaled by 1/sqrt(DH) on load. logD[t,s] = Fc[t] + G[s] (masked s<=t).
// Running per-row max m, signed row-sum ssum, 16x256 f32 accumulator.
// ---------------------------------------------------------------------------
__global__ __launch_bounds__(128) void k_mlstm(
    const float* __restrict__ q, const float* __restrict__ k,
    const float* __restrict__ v, const float* __restrict__ Fc,
    const float* __restrict__ G, float* __restrict__ hout) {
  __shared__ __bf16 CwLds[4][16][32];
  const int lane = threadIdx.x & 31;
  const int wid = threadIdx.x >> 5;
  const int bh = blockIdx.y;
  const int b = bh >> 2, h = bh & 3;
  const int t0 = blockIdx.x * 64 + wid * 16;
  const int ld = 1024;
  const int hoff = h * 256;
  const int cl = lane & 15;
  const int off = (lane >> 4) << 3;  // 0 or 8
  const float* Fch = Fc + bh * 2048;
  const float* Gh = G + bh * 2048;

  // preload q fragments (A layout), scaled by 1/sqrt(256)
  v16bf qf[8];
  {
    const float* qp = q + (b * 2048 + t0 + cl) * ld + hoff;
#pragma unroll
    for (int dc = 0; dc < 8; ++dc)
      qf[dc] = frag_gf32_scaled(qp + dc * 32 + off, 0.0625f);
  }

  v8f acc[16];
#pragma unroll
  for (int ti = 0; ti < 16; ++ti) acc[ti] = v8f_zero();

  float mrow[8], ssum[8], fct[8];
#pragma unroll
  for (int r = 0; r < 8; ++r) {
    mrow[r] = -1e30f;
    ssum[r] = 0.0f;
    fct[r] = Fch[t0 + r + off];
  }

  for (int s0 = 0; s0 <= t0 + 15; s0 += 32) {
    // ---- scores: two 16x16 subtiles over DH=256 (8 k-chunks) ----
    v8f sc0 = v8f_zero(), sc1 = v8f_zero();
    const float g0 = Gh[s0 + cl];
    const float g1 = Gh[s0 + 16 + cl];
    const float* kp0 = k + (b * 2048 + s0 + cl) * ld + hoff + off;
    const float* kp1 = kp0 + 16 * ld;
#pragma unroll
    for (int dc = 0; dc < 8; ++dc) {
      v16bf kf0 = frag_gf32(kp0 + dc * 32);
      v16bf kf1 = frag_gf32(kp1 + dc * 32);
      sc0 = wmma_bf16(qf[dc], kf0, sc0);
      sc1 = wmma_bf16(qf[dc], kf1, sc1);
    }

    // ---- log-decay weighting + online rescale (per row) ----
    float scale[8], cw0[8], cw1[8];
#pragma unroll
    for (int r = 0; r < 8; ++r) {
      int trow = t0 + r + off;
      float ld0 = fct[r] + g0;
      float ld1 = fct[r] + g1;
      bool v0 = (s0 + cl) <= trow;
      bool v1 = (s0 + 16 + cl) <= trow;
      float lm = fmaxf(v0 ? ld0 : -1e30f, v1 ? ld1 : -1e30f);
#pragma unroll
      for (int d = 1; d < 16; d <<= 1) lm = fmaxf(lm, __shfl_xor(lm, d, 32));
      float mnew = fmaxf(mrow[r], lm);
      scale[r] = __expf(mrow[r] - mnew);
      mrow[r] = mnew;
      float w0 = v0 ? __expf(ld0 - mnew) : 0.0f;
      float w1 = v1 ? __expf(ld1 - mnew) : 0.0f;
      cw0[r] = sc0[r] * w0;
      cw1[r] = sc1[r] * w1;
      float rs = cw0[r] + cw1[r];
#pragma unroll
      for (int d = 1; d < 16; d <<= 1) rs += __shfl_xor(rs, d, 32);
      ssum[r] = ssum[r] * scale[r] + rs;
    }

    // rescale accumulator rows
#pragma unroll
    for (int ti = 0; ti < 16; ++ti)
#pragma unroll
      for (int r = 0; r < 8; ++r) acc[ti][r] *= scale[r];

    // ---- stage weighted scores to LDS (D layout -> A layout) ----
#pragma unroll
    for (int r = 0; r < 8; ++r) {
      CwLds[wid][r + off][cl] = (__bf16)cw0[r];
      CwLds[wid][r + off][16 + cl] = (__bf16)cw1[r];
    }
    asm volatile("s_wait_dscnt 0" ::: "memory");
    v16bf af = frag_lds(&CwLds[wid][cl][off]);

    // ---- acc += Cw(16x32) @ V(32x256) ----
#pragma unroll
    for (int ti = 0; ti < 16; ++ti) {
      v16bf vf;
      const float* vp = v + (b * 2048 + s0 + off) * ld + hoff + ti * 16 + cl;
#pragma unroll
      for (int j = 0; j < 8; ++j) {
        vf[j] = (__bf16)vp[j * ld];
        vf[8 + j] = (__bf16)vp[(16 + j) * ld];
      }
      acc[ti] = wmma_bf16(af, vf, acc[ti]);
    }
  }

  // ---- finalize: n = max(|ssum|, exp(-m)); h = acc / n ----
  float inv[8];
#pragma unroll
  for (int r = 0; r < 8; ++r) {
    float nn = fmaxf(fabsf(ssum[r]), __expf(-mrow[r]));
    inv[r] = 1.0f / nn;
  }
#pragma unroll
  for (int ti = 0; ti < 16; ++ti) {
    float* op = hout + (b * 2048 + t0 + off) * ld + hoff + ti * 16 + cl;
#pragma unroll
    for (int r = 0; r < 8; ++r) op[r * ld] = acc[ti][r] * inv[r];
  }
}

// ---------------------------------------------------------------------------
// Per-head GroupNorm + skip + SiLU(z) gate. One wave per (row, head).
// ---------------------------------------------------------------------------
__global__ __launch_bounds__(256) void k_gn_gate(
    const float* __restrict__ hbuf, const float* __restrict__ xc,
    const float* __restrict__ z, int zld, const float* __restrict__ gng,
    const float* __restrict__ gnb, const float* __restrict__ skip,
    float* __restrict__ hout) {
  const int gi = blockIdx.x * 8 + (threadIdx.x >> 5);  // row*4 + h
  const int row = gi >> 2, h = gi & 3;
  const int lane = threadIdx.x & 31;
  const float* hp = hbuf + row * 1024 + h * 256;
  float vals[8];
  float s = 0.0f;
#pragma unroll
  for (int i = 0; i < 8; ++i) {
    vals[i] = hp[lane + i * 32];
    s += vals[i];
  }
  s = wave_red_sum32(s);
  float mu = s * (1.0f / 256.0f);
  float vs = 0.0f;
#pragma unroll
  for (int i = 0; i < 8; ++i) {
    float d = vals[i] - mu;
    vs += d * d;
  }
  vs = wave_red_sum32(vs);
  float rr = rsqrtf(vs * (1.0f / 256.0f) + 1e-6f);
#pragma unroll
  for (int i = 0; i < 8; ++i) {
    int c = h * 256 + lane + i * 32;
    float hn = (vals[i] - mu) * rr * gng[c] + gnb[c];
    float xcv = xc[row * 1024 + c];
    float zv = z[row * zld + c];
    float sil = zv / (1.0f + __expf(-zv));
    hout[row * 1024 + c] = (hn + skip[c] * xcv) * sil;
  }
}

// ---------------------------------------------------------------------------
// Host-side orchestration
// ---------------------------------------------------------------------------
extern "C" void kernel_launch(void* const* d_in, const int* in_sizes, int n_in,
                              void* d_out, int out_size, void* d_ws,
                              size_t ws_size, hipStream_t stream) {
  (void)in_sizes;
  (void)n_in;
  (void)out_size;
  const int M = 4 * 2048;  // B*S = 8192

  const float* x1 = (const float*)d_in[0];      // (M,256)
  const float* x2 = (const float*)d_in[1];      // (M,512)
  const float* W_init = (const float*)d_in[2];  // (256,512)
  const float* b_init = (const float*)d_in[3];  // (512)
  const float* ln_out_g = (const float*)d_in[36];
  const float* ln_out_b = (const float*)d_in[37];
  float* out = (float*)d_out;

  // workspace layout (floats)
  size_t need = (size_t)M * (512 + 512 + 1024 * 5 + 8) + 2 * 16 * 2048;
  if (ws_size < need * sizeof(float)) return;
  float* ws = (float*)d_ws;
  float* xbuf = ws;            ws += (size_t)M * 512;
  float* xnbuf = ws;           ws += (size_t)M * 512;
  float* xinbuf = ws;          ws += (size_t)M * 1024;  // aliased as hbuf later
  float* zbuf = ws;            ws += (size_t)M * 1024;
  float* xcbuf = ws;           ws += (size_t)M * 1024;
  float* qbuf = ws;            ws += (size_t)M * 1024;  // aliased as hout later
  float* kbuf = ws;            ws += (size_t)M * 1024;
  float* vbuf = ws;            ws += (size_t)M * 1024;
  float* iraw = ws;            ws += (size_t)M * 4;
  float* fraw = ws;            ws += (size_t)M * 4;
  float* FcB = ws;             ws += 16 * 2048;
  float* GB = ws;              ws += 16 * 2048;
  float* hbuf = xinbuf;  // x_in fully consumed before mLSTM writes h
  float* hout = qbuf;    // q fully consumed before gate writes h_out

  // x = x1 @ W_init + b_init + x2
  k_gemm<<<dim3(512 / 128, M / 128), 256, 0, stream>>>(
      x1, 256, W_init, 512, b_init, x2, 512, xbuf, 512, 256);

  for (int bi = 0; bi < 2; ++bi) {
    const int base = 4 + bi * 16;
    const float* ln_g   = (const float*)d_in[base + 0];
    const float* ln_b   = (const float*)d_in[base + 1];
    const float* W_up   = (const float*)d_in[base + 2];  // (512,2048)
    const float* conv_w = (const float*)d_in[base + 3];  // (4,1024)
    const float* conv_b = (const float*)d_in[base + 4];
    const float* Wq     = (const float*)d_in[base + 5];
    const float* Wk     = (const float*)d_in[base + 6];
    const float* Wv     = (const float*)d_in[base + 7];
    const float* Wi     = (const float*)d_in[base + 8];
    const float* bi_p   = (const float*)d_in[base + 9];
    const float* Wf     = (const float*)d_in[base + 10];
    const float* bf_p   = (const float*)d_in[base + 11];
    const float* skip   = (const float*)d_in[base + 12];
    const float* gn_g   = (const float*)d_in[base + 13];
    const float* gn_b   = (const float*)d_in[base + 14];
    const float* W_down = (const float*)d_in[base + 15];  // (1024,512)

    k_ln<<<M / 8, 256, 0, stream>>>(xbuf, ln_g, ln_b, xnbuf);
    // x_in = xn @ W_up[:, :1024] ; z = xn @ W_up[:, 1024:]
    k_gemm<<<dim3(8, M / 128), 256, 0, stream>>>(
        xnbuf, 512, W_up, 2048, nullptr, nullptr, 0, xinbuf, 1024, 512);
    k_gemm<<<dim3(8, M / 128), 256, 0, stream>>>(
        xnbuf, 512, W_up + 1024, 2048, nullptr, nullptr, 0, zbuf, 1024, 512);
    k_conv_silu<<<(M * 1024) / 256, 256, 0, stream>>>(xinbuf, 1024, conv_w,
                                                      conv_b, xcbuf);
    k_gemm<<<dim3(8, M / 128), 256, 0, stream>>>(
        xcbuf, 1024, Wq, 1024, nullptr, nullptr, 0, qbuf, 1024, 1024);
    k_gemm<<<dim3(8, M / 128), 256, 0, stream>>>(
        xcbuf, 1024, Wk, 1024, nullptr, nullptr, 0, kbuf, 1024, 1024);
    k_gemm<<<dim3(8, M / 128), 256, 0, stream>>>(
        xinbuf, 1024, Wv, 1024, nullptr, nullptr, 0, vbuf, 1024, 1024);
    k_ifproj<<<M / 8, 256, 0, stream>>>(xcbuf, Wi, bi_p, Wf, bf_p, iraw, fraw);
    k_scan<<<16, 32, 0, stream>>>(iraw, fraw, FcB, GB);
    k_mlstm<<<dim3(2048 / 64, 16), 128, 0, stream>>>(qbuf, kbuf, vbuf, FcB, GB,
                                                     hbuf);
    k_gn_gate<<<(M * 4) / 8, 256, 0, stream>>>(hbuf, xcbuf, zbuf, 1024, gn_g,
                                               gn_b, skip, hout);
    // x = x + h_out @ W_down
    k_gemm<<<dim3(4, M / 128), 256, 0, stream>>>(
        hout, 1024, W_down, 512, nullptr, xbuf, 512, xbuf, 512, 1024);
  }

  k_ln<<<M / 8, 256, 0, stream>>>(xbuf, ln_out_g, ln_out_b, out);
}
